// ISTIncrementalEncoder_74268574483141
// MI455X (gfx1250) — compile-verified, matrix operand order checked by reading, not run on hardware
//
#include <hip/hip_runtime.h>
#include <hip/hip_bf16.h>

// ---------------------------------------------------------------------------
// ISTIncrementalEncoder for MI455X (gfx1250, wave32, WMMA bf16 path)
// B=16384, F=128, D=256, A=256, T=64
// One wave per block, 32 batch rows per wave (two 16-row M tiles sharing
// every WMMA B fragment -> 2x arithmetic intensity on the weight stream).
// ---------------------------------------------------------------------------

typedef __bf16 v16b __attribute__((ext_vector_type(16)));
typedef float  v8f  __attribute__((ext_vector_type(8)));

#define BB 16384
#define FF 128
#define DD 256
#define AA 256
#define TT 64
#define DECAYF 0.9f

// ws byte offsets
#define WS_HIST 0          // 64*256 f32 = 65536 B
#define WS_TOKP 65536      // 64*256 f32 = 65536 B
#define WS_FRAG 131072     // packed bf16 WMMA B-fragments
// fragment sub-offsets inside WS_FRAG (each fragment = 32 lanes * 32 B = 1 KB)
#define OFF_WIH 0          // 48 ntiles * 4 ktiles  = 192 frags
#define OFF_WHH 196608     // 48 ntiles * 8 ktiles  = 384 frags
#define OFF_WC  589824     // 16 ntiles * 8 ktiles  = 128 frags
#define OFF_HIST 720896    // 16 ntiles * 2 ktiles  =  32 frags

__device__ __forceinline__ float sigm_f(float x)  { return 1.0f / (1.0f + __expf(-x)); }
__device__ __forceinline__ float tanh_f(float x)  { return 1.0f - 2.0f / (1.0f + __expf(2.0f * x)); }

// k-index inside a 32-wide K tile for the 16-bit wave32 WMMA A/B register
// layout (ISA 7.12.2): VGPR v, half h, lane-group g.
__device__ __forceinline__ int klocal(int s, int g) {
    int v = s >> 1, h = s & 1;
    return ((v < 4) ? (2 * v + h) : (16 + 2 * (v - 4) + h)) + g * 8;
}

__device__ __forceinline__ v8f wmma_bf16(v16b a, v16b b, v8f c) {
    return __builtin_amdgcn_wmma_f32_16x16x32_bf16(false, a, false, b, (short)0, c,
                                                   false, false);
}

// ---------------------------------------------------------------------------
// Prep kernel 1: hist = tanh(tokens), elementwise over T*D = 16384
// ---------------------------------------------------------------------------
__global__ void k_hist(const float* __restrict__ tok, float* __restrict__ hist) {
    int i = blockIdx.x * blockDim.x + threadIdx.x;
    hist[i] = tanh_f(tok[i]);
}

// ---------------------------------------------------------------------------
// Prep kernel 2: tok_proj[t][a] = bh[a] + sum_d hist[t][d] * Wh[a][d]
// ---------------------------------------------------------------------------
__global__ void k_tokproj(const float* __restrict__ hist, const float* __restrict__ Wh,
                          const float* __restrict__ bh, float* __restrict__ tokp) {
    int i = blockIdx.x * blockDim.x + threadIdx.x;   // 64*256 outputs
    int t = i >> 8, a = i & 255;
    const float* hrow = hist + t * DD;
    const float* wrow = Wh + a * DD;
    float s = bh[a];
#pragma unroll 4
    for (int d = 0; d < DD; ++d) s += hrow[d] * wrow[d];
    tokp[i] = s;
}

// ---------------------------------------------------------------------------
// Prep kernel 3: pack all WMMA B-matrix fragments as bf16, one wave per frag.
// B(k,n) = W[n][k] for transposed weights, B(k,n) = src[k][n] for hist.
// Fragment storage: frag*1KB + lane*32B -> one v16b per lane per fragment.
// ---------------------------------------------------------------------------
__global__ void k_pack(const float* __restrict__ Wih, const float* __restrict__ Whh,
                       const float* __restrict__ Wc, const float* __restrict__ hist,
                       char* __restrict__ frags) {
    int lane = threadIdx.x & 31;
    int f = blockIdx.x * 8 + (threadIdx.x >> 5);     // 0..735
    if (f >= 736) return;

    const float* src;
    int ld, nt, kt;
    bool trans = false;
    v16b* dst;
    if (f < 192) {                                    // W_ih^T : K=128(4), N=768(48)
        nt = f / 4; kt = f % 4; src = Wih; ld = FF;
        dst = (v16b*)(frags + OFF_WIH) + f * 32;
    } else if (f < 576) {                             // W_hh^T : K=256(8), N=768(48)
        int f2 = f - 192;
        nt = f2 / 8; kt = f2 % 8; src = Whh; ld = DD;
        dst = (v16b*)(frags + OFF_WHH) + f2 * 32;
    } else if (f < 704) {                             // W_c^T : K=256(8), N=256(16)
        int f2 = f - 576;
        nt = f2 / 8; kt = f2 % 8; src = Wc; ld = DD;
        dst = (v16b*)(frags + OFF_WC) + f2 * 32;
    } else {                                          // hist : K=64(2), N=256(16)
        int f2 = f - 704;
        nt = f2 / 2; kt = f2 % 2; src = hist; ld = DD; trans = true;
        dst = (v16b*)(frags + OFF_HIST) + f2 * 32;
    }

    int n = nt * 16 + (lane & 15);
    int g = lane >> 4;
    v16b val;
#pragma unroll
    for (int s = 0; s < 16; ++s) {
        int k = kt * 32 + klocal(s, g);
        float x = trans ? src[k * ld + n] : src[n * ld + k];
        val[s] = (__bf16)x;
    }
    dst[lane] = val;
}

// ---------------------------------------------------------------------------
// Main fused kernel: 1 wave per block, 32 rows per wave (2 x 16-row M tiles).
// ---------------------------------------------------------------------------
__global__ __launch_bounds__(32) void ist_main(
    const float* __restrict__ feat, const float* __restrict__ hin,
    const float* __restrict__ accu, const int* __restrict__ mask,
    const float* __restrict__ b_ih, const float* __restrict__ b_hh,
    const float* __restrict__ bc, const float* __restrict__ vvec,
    const float* __restrict__ bv, const float* __restrict__ tokp,
    const char* __restrict__ frags, float* __restrict__ out)
{
    __shared__ float buf[2][16 * DD];   // per-M-tile: hnew, then ctx (32 KB)
    __shared__ float wgt[2][16 * TT];   // scores -> softmax weights (8 KB)
    __shared__ float vsh[AA];           // attention v (1 KB)

    const int lane = threadIdx.x;
    const int m16 = lane & 15, g = lane >> 4;
    const int mrow = g ? 8 : 0;          // C/D layout: lanes 16-31 hold M = i+8
    const int r0 = blockIdx.x * 32;

    // Warm L2/L0 for the accumulator tile consumed in Phase E (global_prefetch_b8).
    __builtin_prefetch(accu + (size_t)(r0 + lane) * DD, 0, 1);
    __builtin_prefetch(accu + (size_t)(r0 + lane) * DD + 128, 0, 1);

    for (int i = lane; i < AA; i += 32) vsh[i] = vvec[i];
    __syncthreads();

    const v16b* fWih = (const v16b*)(frags + OFF_WIH);
    const v16b* fWhh = (const v16b*)(frags + OFF_WHH);
    const v16b* fWc  = (const v16b*)(frags + OFF_WC);
    const v16b* fHis = (const v16b*)(frags + OFF_HIST);

    // ---- Phase A: gather A fragments (features 32x128, h 32x256) as bf16 ----
    v16b fa[2][4], fh[2][8];
#pragma unroll
    for (int u = 0; u < 2; ++u) {
        const float* frow = feat + (size_t)(r0 + u * 16 + m16) * FF;
#pragma unroll
        for (int kt = 0; kt < 4; ++kt)
#pragma unroll
            for (int p = 0; p < 8; ++p) {
                int k = kt * 32 + klocal(2 * p, g);
                float2 xy = *(const float2*)(frow + k);
                fa[u][kt][2 * p]     = (__bf16)xy.x;
                fa[u][kt][2 * p + 1] = (__bf16)xy.y;
            }
        const float* hrow = hin + (size_t)(r0 + u * 16 + m16) * DD;
#pragma unroll
        for (int kt = 0; kt < 8; ++kt)
#pragma unroll
            for (int p = 0; p < 8; ++p) {
                int k = kt * 32 + klocal(2 * p, g);
                float2 xy = *(const float2*)(hrow + k);
                fh[u][kt][2 * p]     = (__bf16)xy.x;
                fh[u][kt][2 * p + 1] = (__bf16)xy.y;
            }
    }

    // ---- Phase B: GRU. Every B fragment loaded once, used by both M tiles ----
#pragma unroll 1
    for (int j = 0; j < 16; ++j) {
        v8f air[2] = {}, aiz[2] = {}, ain_[2] = {};
        v8f ahr[2] = {}, ahz[2] = {}, ahn[2] = {};
#pragma unroll
        for (int kt = 0; kt < 4; ++kt) {
            v16b br = fWih[((j)      * 4 + kt) * 32 + lane];
            v16b bz = fWih[((16 + j) * 4 + kt) * 32 + lane];
            v16b bn = fWih[((32 + j) * 4 + kt) * 32 + lane];
#pragma unroll
            for (int u = 0; u < 2; ++u) {
                air[u]  = wmma_bf16(fa[u][kt], br, air[u]);
                aiz[u]  = wmma_bf16(fa[u][kt], bz, aiz[u]);
                ain_[u] = wmma_bf16(fa[u][kt], bn, ain_[u]);
            }
        }
#pragma unroll
        for (int kt = 0; kt < 8; ++kt) {
            v16b br = fWhh[((j)      * 8 + kt) * 32 + lane];
            v16b bz = fWhh[((16 + j) * 8 + kt) * 32 + lane];
            v16b bn = fWhh[((32 + j) * 8 + kt) * 32 + lane];
#pragma unroll
            for (int u = 0; u < 2; ++u) {
                ahr[u] = wmma_bf16(fh[u][kt], br, ahr[u]);
                ahz[u] = wmma_bf16(fh[u][kt], bz, ahz[u]);
                ahn[u] = wmma_bf16(fh[u][kt], bn, ahn[u]);
            }
        }
        const int n = j * 16 + m16;
        const float bir = b_ih[n], biz = b_ih[DD + n], bin = b_ih[2 * DD + n];
        const float bhr = b_hh[n], bhz = b_hh[DD + n], bhn = b_hh[2 * DD + n];
#pragma unroll
        for (int u = 0; u < 2; ++u)
#pragma unroll
            for (int i = 0; i < 8; ++i) {
                int m = i + mrow;
                int row = r0 + u * 16 + m;
                float r  = sigm_f((air[u][i] + bir) + (ahr[u][i] + bhr));
                float z  = sigm_f((aiz[u][i] + biz) + (ahz[u][i] + bhz));
                float nn = tanh_f((ain_[u][i] + bin) + r * (ahn[u][i] + bhn));
                float hold = hin[(size_t)row * DD + n];
                float hnew = (1.0f - z) * nn + z * hold;
                out[(size_t)BB * DD + (size_t)row * DD + n] = mask[row] ? hnew : hold;
                buf[u][m * DD + n] = hnew;   // unmasked h_new feeds attention
            }
    }
    __syncthreads();

    // ---- Phase C: h_new -> A fragments, ctx = h_new * Wc^T + bc ----
    v16b fhn[2][8];
#pragma unroll
    for (int u = 0; u < 2; ++u) {
        const float* brow = &buf[u][m16 * DD];
#pragma unroll
        for (int kt = 0; kt < 8; ++kt)
#pragma unroll
            for (int p = 0; p < 8; ++p) {
                int k = kt * 32 + klocal(2 * p, g);
                float2 xy = *(const float2*)(brow + k);
                fhn[u][kt][2 * p]     = (__bf16)xy.x;
                fhn[u][kt][2 * p + 1] = (__bf16)xy.y;
            }
    }
#pragma unroll 1
    for (int j = 0; j < 16; ++j) {
        v8f c[2] = {};
#pragma unroll
        for (int kt = 0; kt < 8; ++kt) {
            v16b b = fWc[(j * 8 + kt) * 32 + lane];
#pragma unroll
            for (int u = 0; u < 2; ++u)
                c[u] = wmma_bf16(fhn[u][kt], b, c[u]);
        }
        const int n = j * 16 + m16;
        const float b = bc[n];
#pragma unroll
        for (int u = 0; u < 2; ++u)
#pragma unroll
            for (int i = 0; i < 8; ++i)
                buf[u][(i + mrow) * DD + n] = c[u][i] + b;   // ctx overwrites hnew
    }
    __syncthreads();

    // ---- Phase D: scores[m][t] = sum_a v[a]*tanh(tok_proj[t][a]+ctx[m][a]) ----
    {
        const float bv0 = bv[0];
        const float4* vh4 = (const float4*)(vsh + g * 128);
#pragma unroll 1
        for (int u = 0; u < 2; ++u) {
            const float4* cx4 = (const float4*)(&buf[u][m16 * DD + g * 128]);
#pragma unroll 1
            for (int t = 0; t < TT; ++t) {
                const float4* tp4 = (const float4*)(tokp + t * AA + g * 128);
                float s0 = 0.f, s1 = 0.f, s2 = 0.f, s3 = 0.f;
#pragma unroll 4
                for (int q = 0; q < 32; ++q) {
                    float4 tv = tp4[q], cv = cx4[q], vv = vh4[q];
                    s0 += vv.x * tanh_f(tv.x + cv.x);
                    s1 += vv.y * tanh_f(tv.y + cv.y);
                    s2 += vv.z * tanh_f(tv.z + cv.z);
                    s3 += vv.w * tanh_f(tv.w + cv.w);
                }
                float s = (s0 + s1) + (s2 + s3);
                s += __shfl_xor(s, 16, 32);
                if (g == 0) wgt[u][m16 * TT + t] = s + bv0;
            }
        }
    }
    __syncthreads();

    // ---- softmax over T=64 per row (two M tiles) ----
#pragma unroll 1
    for (int u = 0; u < 2; ++u) {
        float mx = -1e30f;
        for (int t = g * 32; t < g * 32 + 32; ++t) mx = fmaxf(mx, wgt[u][m16 * TT + t]);
        mx = fmaxf(mx, __shfl_xor(mx, 16, 32));
        float sum = 0.0f;
        for (int t = g * 32; t < g * 32 + 32; ++t) {
            float e = __expf(wgt[u][m16 * TT + t] - mx);
            wgt[u][m16 * TT + t] = e;
            sum += e;
        }
        sum += __shfl_xor(sum, 16, 32);
        float inv = 1.0f / sum;
        for (int t = g * 32; t < g * 32 + 32; ++t) wgt[u][m16 * TT + t] *= inv;
    }
    __syncthreads();

    // ---- Phase E: emb = weights * hist; masked decayed accumulator update ----
    v16b aw[2][2];
#pragma unroll
    for (int u = 0; u < 2; ++u)
#pragma unroll
        for (int kt = 0; kt < 2; ++kt)
#pragma unroll
            for (int s = 0; s < 16; ++s) {
                int k = kt * 32 + klocal(s, g);
                aw[u][kt][s] = (__bf16)wgt[u][m16 * TT + k];
            }
#pragma unroll 1
    for (int j = 0; j < 16; ++j) {
        v16b b0 = fHis[(j * 2 + 0) * 32 + lane];
        v16b b1 = fHis[(j * 2 + 1) * 32 + lane];
        const int n = j * 16 + m16;
#pragma unroll
        for (int u = 0; u < 2; ++u) {
            v8f c = {};
            c = wmma_bf16(aw[u][0], b0, c);
            c = wmma_bf16(aw[u][1], b1, c);
#pragma unroll
            for (int i = 0; i < 8; ++i) {
                int row = r0 + u * 16 + i + mrow;
                float aold = accu[(size_t)row * DD + n];
                out[(size_t)row * DD + n] = mask[row] ? (aold * DECAYF + c[i]) : aold;
            }
        }
    }
}

// ---------------------------------------------------------------------------
extern "C" void kernel_launch(void* const* d_in, const int* in_sizes, int n_in,
                              void* d_out, int out_size, void* d_ws, size_t ws_size,
                              hipStream_t stream) {
    (void)in_sizes; (void)n_in; (void)out_size; (void)ws_size;
    const float* feat = (const float*)d_in[0];
    const float* h    = (const float*)d_in[1];
    const float* accu = (const float*)d_in[2];
    const int*   mask = (const int*)d_in[3];
    const float* tok  = (const float*)d_in[4];
    const float* Wih  = (const float*)d_in[5];
    const float* Whh  = (const float*)d_in[6];
    const float* bih  = (const float*)d_in[7];
    const float* bhh  = (const float*)d_in[8];
    const float* Wh   = (const float*)d_in[9];
    const float* bh   = (const float*)d_in[10];
    const float* Wc   = (const float*)d_in[11];
    const float* bc   = (const float*)d_in[12];
    const float* v    = (const float*)d_in[13];
    const float* bv   = (const float*)d_in[14];
    float* out = (float*)d_out;
    char*  ws  = (char*)d_ws;

    float* hist = (float*)(ws + WS_HIST);
    float* tokp = (float*)(ws + WS_TOKP);
    char*  frag = ws + WS_FRAG;

    k_hist<<<TT * DD / 256, 256, 0, stream>>>(tok, hist);
    k_tokproj<<<TT * AA / 256, 256, 0, stream>>>(hist, Wh, bh, tokp);
    k_pack<<<92, 256, 0, stream>>>(Wih, Whh, Wc, hist, frag);
    ist_main<<<BB / 32, 32, 0, stream>>>(feat, h, accu, mask, bih, bhh, bc, v, bv,
                                         tokp, frag, out);
}
